// VectorQuantizerEMA_32323923869719
// MI455X (gfx1250) — matrix operand order repaired and checked by reading, not compiled, not run on hardware
//
#include <hip/hip_runtime.h>

// ---------------------------------------------------------------------------
// VQ-VAE EMA vector quantizer for MI455X (gfx1250, wave32).
//   N = 65536 rows, D = 64, K = 1024 codes.
//   Distance GEMM (-2 x.e^T) via v_wmma_f32_16x16x32_bf16 (fp32 accum).
//   Codebook staged into LDS with global_load_async_to_lds_b128 (ASYNCcnt),
//   consumed via ds_load_b128 inside the WMMA loop.
// ---------------------------------------------------------------------------

#define K_CODES 1024
#define D_DIM   64
#define N_VEC   65536
#define DECAY   0.99f
#define EPSF    1e-5f

typedef __attribute__((ext_vector_type(16))) __bf16 v16bf;
typedef __attribute__((ext_vector_type(8)))  float  v8f;

union BfVec { v16bf v; unsigned short u[16]; };

__device__ __forceinline__ unsigned short f2bf(float f) {
  // round-to-nearest-even fp32 -> bf16
  unsigned int u = __float_as_uint(f);
  unsigned int r = u + 0x7FFFu + ((u >> 16) & 1u);
  return (unsigned short)(r >> 16);
}

// Low 32 bits of a generic pointer to LDS == LDS byte offset (ISA: flat->LDS
// truncates to addr[31:0]).
__device__ __forceinline__ unsigned lds_off(const void* p) {
  return (unsigned)(uintptr_t)p;
}

// Workspace layout (bytes):
//   [0,       131072)  Bpack : bf16 codebook in WMMA B per-lane layout
//   [131072,  135168)  norm_e: |e_k|^2, K floats
//   [135168,  139264)  counts: K floats (zeroed per call)
//   [139264,  401408)  dw    : K*D floats (zeroed per call)
//   [401408,  401412)  loss accumulator (zeroed per call)
#define WS_BPACK  0
#define WS_NORM   131072
#define WS_COUNTS 135168
#define WS_DW     139264
#define WS_LOSS   401408

// --- pack codebook into per-lane WMMA B operands (contiguous k) --------------
// Bpack element index = ((cb*2 + kb)*32 + lane)*16 + i
//   lane 0..15  hold K = kb*32 + 0..15  (2 per VGPR)
//   lane 16..31 hold K = kb*32 + 16..31
__global__ void vq_prep_pack(const float* __restrict__ embed,
                             unsigned short* __restrict__ bpack) {
  int tid  = blockIdx.x * blockDim.x + threadIdx.x;   // 0..4095
  int lane = tid & 31;
  int pair = tid >> 5;            // 0..127
  int cb   = pair >> 1;           // code block 0..63
  int kb   = pair & 1;            // K half (dims 0..31 / 32..63)
  int n    = lane & 15;
  int hi   = lane >> 4;
  int code = cb * 16 + n;
  unsigned short* dst = bpack + ((size_t)(cb * 64 + kb * 32 + lane)) * 16;
  const float* src = embed + (size_t)code * D_DIM + kb * 32 + hi * 16;
#pragma unroll
  for (int i = 0; i < 16; ++i) dst[i] = f2bf(src[i]);
}

// --- |e_k|^2 ----------------------------------------------------------------
__global__ void vq_prep_norm(const float* __restrict__ embed,
                             float* __restrict__ norme) {
  int k = blockIdx.x * blockDim.x + threadIdx.x;      // 0..1023
  const float* row = embed + (size_t)k * D_DIM;
  float s = 0.f;
#pragma unroll
  for (int d = 0; d < D_DIM; ++d) s += row[d] * row[d];
  norme[k] = s;
}

// --- main: distances via WMMA from LDS-staged codebook ----------------------
__global__ void __launch_bounds__(256)
vq_main(const float* __restrict__ zf, const float* __restrict__ embed,
        const unsigned short* __restrict__ bpack,
        const float* __restrict__ norme,
        float* __restrict__ out_q, float* __restrict__ out_idx,
        float* __restrict__ counts, float* __restrict__ dw,
        float* __restrict__ loss_acc) {
  // One 64 KB half of the packed codebook (32 code blocks) + norms + idx slots
  __shared__ __align__(16) unsigned short sB[32 * 64 * 16];  // 65536 B
  __shared__ __align__(16) float sN[K_CODES];                // 4096 B
  __shared__ int s_idx[8 * 16];

  const int tid  = threadIdx.x;
  const int lane = tid & 31;
  const int wave = tid >> 5;
  const int tile = blockIdx.x * 8 + wave;
  const int row0 = tile * 16;
  const int row  = lane & 15;
  const int hi   = lane >> 4;

  // Stage |e|^2 into LDS: 256 threads x 16 B (async; completion covered by the
  // first s_wait_asynccnt below).
  {
    unsigned ldsn = lds_off(sN) + tid * 16;
    int      voff = tid * 16;
    asm volatile("global_load_async_to_lds_b128 %0, %1, %2 offset:0"
                 :: "v"(ldsn), "v"(voff), "s"(norme) : "memory");
  }

  // Pack A operand (contiguous k order, matching Bpack; two K-halves for D=64):
  //   hi=0: V0..7 = K0..15
  //   hi=1: V0..7 = K16..31
  BfVec a0, a1;
  const float* zrow = zf + (size_t)(row0 + row) * D_DIM;
#pragma unroll
  for (int i = 0; i < 16; ++i) {
    int kl = hi * 16 + i;
    a0.u[i] = f2bf(zrow[kl]);
    a1.u[i] = f2bf(zrow[32 + kl]);
  }

  float runmin[8];
  int   runidx[8];
#pragma unroll
  for (int r = 0; r < 8; ++r) { runmin[r] = 3.4e38f; runidx[r] = 0; }

  const unsigned ldsb = lds_off(sB) + tid * 256;

  for (int h = 0; h < 2; ++h) {
    if (h) __syncthreads();   // all waves done consuming previous half
    // Stage 64 KB of Bpack: 256 threads x 256 B = 16 async b128 each.
    // offset: advances both the LDS and global addresses (ISA pseudocode).
    {
      int voff = h * 65536 + tid * 256;
#pragma unroll
      for (int i = 0; i < 16; ++i) {
        asm volatile("global_load_async_to_lds_b128 %0, %1, %2 offset:%3"
                     :: "v"(ldsb), "v"(voff), "s"(bpack), "n"(i * 16)
                     : "memory");
      }
    }
    asm volatile("s_wait_asynccnt 0x0" ::: "memory");
    __syncthreads();          // staged data visible to all waves

#pragma unroll 4
    for (int cbl = 0; cbl < 32; ++cbl) {
      const int cb = h * 32 + cbl;
      const unsigned short* pb = sB + ((size_t)(cbl * 64 + lane)) * 16;
      BfVec b0, b1;
      b0.v = *(const v16bf*)pb;          // kb = 0 (dims 0..31)
      b1.v = *(const v16bf*)(pb + 512);  // kb = 1 (dims 32..63)

      v8f c = {};
      c = __builtin_amdgcn_wmma_f32_16x16x32_bf16(false, a0.v, false, b0.v,
                                                  (short)0, c, false, false);
      c = __builtin_amdgcn_wmma_f32_16x16x32_bf16(false, a1.v, false, b1.v,
                                                  (short)0, c, false, false);

      const int code = cb * 16 + (lane & 15);   // column N of this lane
      const float nv = sN[code];
#pragma unroll
      for (int r = 0; r < 8; ++r) {
        float s = nv - 2.0f * c[r];             // |e|^2 - 2 x.e (argmin-equiv)
        if (s < runmin[r]) { runmin[r] = s; runidx[r] = code; }
      }
    }
  }

  // Butterfly min-reduce across the 16 lanes of each half-wave.
  // Lanes 0..15 hold rows 0..7; lanes 16..31 hold rows 8..15.
#pragma unroll
  for (int m = 1; m <= 8; m <<= 1) {
#pragma unroll
    for (int r = 0; r < 8; ++r) {
      float ov = __shfl_xor(runmin[r], m, 32);
      int   oi = __shfl_xor(runidx[r], m, 32);
      if (ov < runmin[r] || (ov == runmin[r] && oi < runidx[r])) {
        runmin[r] = ov; runidx[r] = oi;
      }
    }
  }

  if (lane == 0 || lane == 16) {
#pragma unroll
    for (int r = 0; r < 8; ++r) s_idx[wave * 16 + hi * 8 + r] = runidx[r];
  }
  __syncthreads();

  // Each lane handles half a row: row rl = lane>>1, dims [half*32, half*32+32).
  const int rl   = lane >> 1;
  const int half = lane & 1;
  const int code = s_idx[wave * 16 + rl];
  const int grow = row0 + rl;
  const float* erow = embed + (size_t)code * D_DIM + half * 32;
  const float* zr   = zf   + (size_t)grow * D_DIM + half * 32;
  float* qout = out_q + (size_t)grow * D_DIM + half * 32;
  float* dwp  = dw    + (size_t)code * D_DIM + half * 32;

  float lsum = 0.f;
#pragma unroll
  for (int d = 0; d < 32; ++d) {
    float e = erow[d];
    float z = zr[d];
    qout[d] = e;                       // quantized_st == quantized numerically
    float df = e - z;
    lsum += df * df;
    atomicAdd(&dwp[d], z);             // segment_sum(flat)
  }
  if (half == 0) {
    atomicAdd(&counts[code], 1.0f);    // segment_sum(ones)
    out_idx[grow] = (float)code;       // indices emitted as float in d_out
  }

  // loss = 1.25 * mean((q-z)^2); accumulate via atomics
#pragma unroll
  for (int m = 1; m < 32; m <<= 1) lsum += __shfl_xor(lsum, m, 32);
  if (lane == 0) atomicAdd(loss_acc, lsum);
}

// --- finalize: EMA updates, normalization, loss -----------------------------
__global__ void __launch_bounds__(1024)
vq_finalize(const float* __restrict__ ema_count,
            const float* __restrict__ ema_weight,
            const float* __restrict__ counts,
            const float* __restrict__ dw,
            const float* __restrict__ loss_acc,
            float* __restrict__ out_loss, float* __restrict__ out_embed,
            float* __restrict__ out_cnt,  float* __restrict__ out_emaw) {
  __shared__ float red[1024];
  const int t = threadIdx.x;          // one thread per code, K = 1024
  float cn = DECAY * ema_count[t] + (1.0f - DECAY) * counts[t];
  red[t] = cn;
  __syncthreads();
  for (int s = 512; s > 0; s >>= 1) {
    if (t < s) red[t] += red[t + s];
    __syncthreads();
  }
  const float n = red[0];
  const float w = (cn + EPSF) / (n + K_CODES * EPSF) * n;
  const float inv = 1.0f / w;
  out_cnt[t] = cn;
  const float* ew_in = ema_weight + (size_t)t * D_DIM;
  const float* dw_in = dw         + (size_t)t * D_DIM;
  float* ew_out = out_emaw  + (size_t)t * D_DIM;
  float* eb_out = out_embed + (size_t)t * D_DIM;
#pragma unroll
  for (int d = 0; d < D_DIM; ++d) {
    float ew = DECAY * ew_in[d] + (1.0f - DECAY) * dw_in[d];
    ew_out[d] = ew;
    eb_out[d] = ew * inv;
  }
  if (t == 0)
    out_loss[0] = 1.25f * loss_acc[0] / (float)(N_VEC * D_DIM);
}

extern "C" void kernel_launch(void* const* d_in, const int* in_sizes, int n_in,
                              void* d_out, int out_size, void* d_ws, size_t ws_size,
                              hipStream_t stream) {
  (void)in_sizes; (void)n_in; (void)out_size; (void)ws_size;
  const float* z_e        = (const float*)d_in[0];
  const float* embed      = (const float*)d_in[1];
  const float* ema_count  = (const float*)d_in[2];
  const float* ema_weight = (const float*)d_in[3];

  char* ws = (char*)d_ws;
  unsigned short* bpack = (unsigned short*)(ws + WS_BPACK);
  float* norme    = (float*)(ws + WS_NORM);
  float* counts   = (float*)(ws + WS_COUNTS);
  float* dw       = (float*)(ws + WS_DW);
  float* loss_acc = (float*)(ws + WS_LOSS);

  // d_out: quantized_st | loss | indices | new_embed | new_ema_count | new_ema_weight
  float* out       = (float*)d_out;
  float* out_q     = out;                       // 4194304
  float* out_loss  = out + 4194304;             // 1
  float* out_idx   = out + 4194305;             // 65536
  float* out_embed = out + 4259841;             // 65536
  float* out_cnt   = out + 4325377;             // 1024
  float* out_emaw  = out + 4326401;             // 65536

  // zero counts + dw + loss (contiguous region)
  hipMemsetAsync(ws + WS_COUNTS, 0, 4096 + 262144 + 4, stream);

  vq_prep_pack<<<16, 256, 0, stream>>>(embed, bpack);
  vq_prep_norm<<<4, 256, 0, stream>>>(embed, norme);
  vq_main<<<N_VEC / 16 / 8, 256, 0, stream>>>(z_e, embed, bpack, norme,
                                              out_q, out_idx, counts, dw, loss_acc);
  vq_finalize<<<1, 1024, 0, stream>>>(ema_count, ema_weight, counts, dw, loss_acc,
                                      out_loss, out_embed, out_cnt, out_emaw);
}
